// GeoGraphBlock_3925600109164
// MI455X (gfx1250) — matrix operand order repaired and checked by reading, not compiled, not run on hardware
//
#include <hip/hip_runtime.h>
#include <hip/hip_bf16.h>
#include <math.h>

// ---------------------------------------------------------------------------
// GeoGraphBlock (GATv2 x5 + pool + BN) for MI455X (gfx1250, wave32, WMMA).
// GEMMs: V_WMMA_F32_16X16X32_BF16 (bf16 in, f32 accumulate), zero-padded
// staging buffers so the inner loop is branch-free, and
// GLOBAL_LOAD_ASYNC_TO_LDS_B128 for tile staging (ASYNCcnt-tracked).
// ---------------------------------------------------------------------------

#define N_NODES   50000
#define N_EDGES   200000
#define N_ANGLES  300000
#define N_GRAPHS  1024
#define HIDC      128          // per-head channels

typedef __attribute__((ext_vector_type(16))) __bf16 v16bf;
typedef __attribute__((ext_vector_type(8)))  __bf16 v8bf;
typedef __attribute__((ext_vector_type(8)))  float  v8f;
typedef __attribute__((ext_vector_type(4)))  int    i32x4;

#ifndef __has_builtin
#define __has_builtin(x) 0
#endif
#if __has_builtin(__builtin_amdgcn_global_load_async_to_lds_b128)
#define HAVE_ASYNC_LDS 1
#else
#define HAVE_ASYNC_LDS 0
#endif

// ----------------------------- helpers -------------------------------------

__device__ __forceinline__ unsigned short f32_to_bf16_bits(float f) {
    unsigned u = __float_as_uint(f);
    unsigned rounding = 0x7FFFu + ((u >> 16) & 1u);   // round to nearest even
    return (unsigned short)((u + rounding) >> 16);
}

__device__ __forceinline__ void atomicMaxF(float* addr, float v) {
    if (v >= 0.0f) atomicMax((int*)addr, __float_as_int(v));
    else           atomicMin((unsigned int*)addr, __float_as_uint(v));
}

// 16B global->LDS copy: async DMA path on gfx1250 when the builtin exists.
__device__ __forceinline__ void cp16_to_lds(const unsigned short* g, unsigned short* l) {
#if HAVE_ASYNC_LDS
    __builtin_amdgcn_global_load_async_to_lds_b128(
        (__attribute__((address_space(1))) i32x4*)g,
        (__attribute__((address_space(3))) i32x4*)l, 0, 0);
#else
    *reinterpret_cast<uint4*>(l) = *reinterpret_cast<const uint4*>(g);
#endif
}

__device__ __forceinline__ void async_wait_all() {
#if HAVE_ASYNC_LDS
#  if __has_builtin(__builtin_amdgcn_s_wait_asynccnt)
    __builtin_amdgcn_s_wait_asynccnt(0);
#  else
    asm volatile("s_wait_asynccnt 0x0" ::: "memory");
#  endif
#endif
}

// ----------------------------- tiny kernels ---------------------------------

__global__ void fill_kernel(float* __restrict__ p, float v, long long n) {
    long long i = (long long)blockIdx.x * blockDim.x + threadIdx.x;
    if (i < n) p[i] = v;
}

// in [M,K] f32 -> out [Mpad,Kpad] bf16, zero padding (total = Mpad*Kpad)
__global__ void convert_pad_kernel(const float* __restrict__ in,
                                   unsigned short* __restrict__ out,
                                   int M, int K, int Kpad, long long total) {
    long long i = (long long)blockIdx.x * blockDim.x + threadIdx.x;
    if (i < total) {
        int m = (int)(i / Kpad), k = (int)(i - (long long)m * Kpad);
        out[i] = (m < M && k < K) ? f32_to_bf16_bits(in[(long long)m * K + k])
                                  : (unsigned short)0;
    }
}

// W [K,Nc] f32 -> Wt [Nc,Kpad] bf16 transposed, zero padding in K
__global__ void wt_convert_pad_kernel(const float* __restrict__ W,
                                      unsigned short* __restrict__ Wt,
                                      int K, int Kpad, int Nc) {
    int idx = blockIdx.x * blockDim.x + threadIdx.x;
    if (idx < Nc * Kpad) {
        int n = idx / Kpad, k = idx - n * Kpad;
        Wt[idx] = (k < K) ? f32_to_bf16_bits(W[k * Nc + n]) : (unsigned short)0;
    }
}

// out[i] = relu?(out[i] + bias[i % HCs])
__global__ void finalize_kernel(float* __restrict__ out, const float* __restrict__ bias,
                                int HCs, long long n, int relu) {
    long long i = (long long)blockIdx.x * blockDim.x + threadIdx.x;
    if (i < n) {
        int j = (int)(i % HCs);
        float v = out[i] + bias[j];
        if (relu) v = fmaxf(v, 0.0f);
        out[i] = v;
    }
}

// ----------------------------- WMMA GEMM ------------------------------------
// C[Mpad,Nc] = A[Mpad,Kpad](bf16) * Wt[Nc,Kpad](bf16)^T (+bias), f32 accum.
// Block: 256 threads (8 waves), tile 64(M) x 128(N), K-step 32.
// Each wave owns a 32x32 output tile = 2x2 WMMA 16x16 fragments.
// All loads/stores are branch-free thanks to padded buffers.
__global__ __launch_bounds__(256) void gemm_bf16_wmma(
    const unsigned short* __restrict__ A,   // [Mpad,Kpad] bf16 bits
    const unsigned short* __restrict__ Bt,  // [Nc,Kpad]  bf16 bits
    const float* __restrict__ bias,         // [Nc] or nullptr
    float* __restrict__ C,                  // [Mpad,Nc] f32
    int Kpad, int Nc)
{
    __shared__ __align__(16) unsigned short As[64 * 32];
    __shared__ __align__(16) unsigned short Bs[128 * 32];

    const int tid  = threadIdx.x;
    const int m0   = blockIdx.x * 64;
    const int n0   = blockIdx.y * 128;
    const int wave = tid >> 5, lane = tid & 31;
    const int wm   = wave >> 2, wn = wave & 3;     // 2x4 wave grid of 32x32 tiles
    const int half = lane >> 4, lr = lane & 15;

    // staging addresses (fixed per thread)
    const int ar = tid >> 2, ac = (tid & 3) * 8;          // A: 64 rows x 32 k
    const int br = tid >> 1, bc = (tid & 1) * 16;         // B: 128 rows x 32 k
    const unsigned short* gA = A  + (long long)(m0 + ar) * Kpad + ac;
    const unsigned short* gB = Bt + (long long)(n0 + br) * Kpad + bc;
    unsigned short* lA  = &As[ar * 32 + ac];
    unsigned short* lB0 = &Bs[br * 32 + bc];
    unsigned short* lB1 = &Bs[br * 32 + bc + 8];

    v8f acc[2][2] = {};

    for (int k0 = 0; k0 < Kpad; k0 += 32) {
        cp16_to_lds(gA + k0, lA);
        cp16_to_lds(gB + k0, lB0);
        cp16_to_lds(gB + k0 + 8, lB1);
        async_wait_all();
        __syncthreads();

        // --- build fragments per CDNA5 16-bit WMMA layouts ---
        // A 16x32: lane(half,m): k = half*8 + i (i<8), 16 + half*8 + (i-8)
        v16bf afrag[2], bfrag[2];
        #pragma unroll
        for (int mi = 0; mi < 2; ++mi) {
            int row = wm * 32 + mi * 16 + lr;
            v8bf lo = *reinterpret_cast<const v8bf*>(&As[row * 32 + half * 8]);
            v8bf hi = *reinterpret_cast<const v8bf*>(&As[row * 32 + 16 + half * 8]);
            afrag[mi] = __builtin_shufflevector(lo, hi,
                0,1,2,3,4,5,6,7,8,9,10,11,12,13,14,15);
        }
        // B 32x16: lane(half,n): k = half*8 + i (i<8), 16 + half*8 + (i-8)
        #pragma unroll
        for (int ni = 0; ni < 2; ++ni) {
            int row = wn * 32 + ni * 16 + lr;
            v8bf lo = *reinterpret_cast<const v8bf*>(&Bs[row * 32 + half * 8]);
            v8bf hi = *reinterpret_cast<const v8bf*>(&Bs[row * 32 + 16 + half * 8]);
            bfrag[ni] = __builtin_shufflevector(lo, hi,
                0,1,2,3,4,5,6,7,8,9,10,11,12,13,14,15);
        }
        #pragma unroll
        for (int mi = 0; mi < 2; ++mi)
            #pragma unroll
            for (int ni = 0; ni < 2; ++ni)
                acc[mi][ni] = __builtin_amdgcn_wmma_f32_16x16x32_bf16(
                    false, afrag[mi], false, bfrag[ni],
                    (short)0, acc[mi][ni], false, false);
        __syncthreads();
    }

    // --- epilogue: C/D layout: elem r -> M = half*8 + r, N = lane&15 ---
    #pragma unroll
    for (int mi = 0; mi < 2; ++mi) {
        #pragma unroll
        for (int ni = 0; ni < 2; ++ni) {
            int gcol = n0 + wn * 32 + ni * 16 + lr;
            float badd = bias ? bias[gcol] : 0.0f;
            #pragma unroll
            for (int r = 0; r < 8; ++r) {
                int grow = m0 + wm * 32 + mi * 16 + half * 8 + r;
                C[(long long)grow * Nc + gcol] = acc[mi][ni][r] + badd;
            }
        }
    }
}

// --------------------------- attention kernels ------------------------------
// One wave32 per (edge, head); each lane handles 4 channels of 128.

__global__ __launch_bounds__(256) void edge_logits_kernel(
    const float* __restrict__ XL, const float* __restrict__ XR,
    const float* __restrict__ EA, const int* __restrict__ src,
    const int* __restrict__ dst, const float* __restrict__ att,
    float* __restrict__ logits, float* __restrict__ mmax, int nE, int Hh)
{
    int task = blockIdx.x * 8 + (threadIdx.x >> 5);
    int lane = threadIdx.x & 31;
    if (task >= nE * Hh) return;
    int e = task / Hh, h = task - e * Hh;
    int HCs = Hh * HIDC;
    int s = src[e], d = dst[e];
    const float* xl = XL + (long long)s * HCs + h * HIDC + lane * 4;
    const float* xr = XR + (long long)d * HCs + h * HIDC + lane * 4;
    const float* ea = EA + (long long)e * HCs + h * HIDC + lane * 4;
    const float* at = att + h * HIDC + lane * 4;
    float acc = 0.0f;
    #pragma unroll
    for (int i = 0; i < 4; ++i) {
        float v = xl[i] + xr[i] + ea[i];
        v = (v > 0.0f) ? v : 0.2f * v;        // leaky_relu(0.2)
        acc += v * at[i];
    }
    #pragma unroll
    for (int off = 16; off > 0; off >>= 1) acc += __shfl_xor(acc, off);
    if (lane == 0) {
        logits[task] = acc;
        atomicMaxF(&mmax[(long long)d * Hh + h], acc);
    }
}

__global__ void edge_exp_kernel(
    const float* __restrict__ logits, const int* __restrict__ dst,
    const float* __restrict__ mmax, float* __restrict__ exs,
    float* __restrict__ den, int nE, int Hh)
{
    int t = blockIdx.x * blockDim.x + threadIdx.x;
    if (t < nE * Hh) {
        int e = t / Hh, h = t - e * Hh;
        int d = dst[e];
        float ex = expf(logits[t] - mmax[(long long)d * Hh + h]);
        exs[t] = ex;
        atomicAdd(&den[(long long)d * Hh + h], ex);
    }
}

__global__ __launch_bounds__(256) void edge_scatter_kernel(
    const float* __restrict__ XL, const int* __restrict__ src,
    const int* __restrict__ dst, const float* __restrict__ exs,
    const float* __restrict__ den, float* __restrict__ out, int nE, int Hh)
{
    int task = blockIdx.x * 8 + (threadIdx.x >> 5);
    int lane = threadIdx.x & 31;
    if (task >= nE * Hh) return;
    int e = task / Hh, h = task - e * Hh;
    int HCs = Hh * HIDC;
    int s = src[e], d = dst[e];
    float alpha = exs[task] / fmaxf(den[(long long)d * Hh + h], 1e-16f);
    const float* xl = XL + (long long)s * HCs + h * HIDC + lane * 4;
    float*       o  = out + (long long)d * HCs + h * HIDC + lane * 4;
    #pragma unroll
    for (int i = 0; i < 4; ++i) atomicAdd(o + i, alpha * xl[i]);
}

// --------------------------- pool + batchnorm --------------------------------

__global__ void count_kernel(const int* __restrict__ batch, float* __restrict__ cnt, int n) {
    int i = blockIdx.x * blockDim.x + threadIdx.x;
    if (i < n) atomicAdd(&cnt[batch[i]], 1.0f);
}

__global__ void pool_kernel(const float* __restrict__ H3, const int* __restrict__ batch,
                            float* __restrict__ pooled, int n) {
    long long i = (long long)blockIdx.x * blockDim.x + threadIdx.x;
    if (i < (long long)n * HIDC) {
        int node = (int)(i / HIDC), j = (int)(i - (long long)node * HIDC);
        atomicAdd(&pooled[batch[node] * HIDC + j], H3[i]);
    }
}

__global__ __launch_bounds__(256) void bn_kernel(
    const float* __restrict__ pooled, const float* __restrict__ cnt,
    const float* __restrict__ gamma, const float* __restrict__ beta,
    float* __restrict__ out, int G)
{
    __shared__ float ssum[256], ssq[256];
    int j = blockIdx.x;                    // channel 0..127
    int tid = threadIdx.x;
    float s = 0.0f, s2 = 0.0f;
    for (int g = tid; g < G; g += 256) {
        float val = pooled[g * HIDC + j] / fmaxf(cnt[g], 1.0f);
        s += val; s2 += val * val;
    }
    ssum[tid] = s; ssq[tid] = s2;
    __syncthreads();
    for (int o = 128; o > 0; o >>= 1) {
        if (tid < o) { ssum[tid] += ssum[tid + o]; ssq[tid] += ssq[tid + o]; }
        __syncthreads();
    }
    float mu  = ssum[0] / (float)G;
    float var = ssq[0] / (float)G - mu * mu;
    float inv = rsqrtf(var + 1e-5f);
    for (int g = tid; g < G; g += 256) {
        float val = pooled[g * HIDC + j] / fmaxf(cnt[g], 1.0f);
        out[g * HIDC + j] = (val - mu) * inv * gamma[j] + beta[j];
    }
}

// ----------------------------- host orchestration ----------------------------

static inline dim3 g1(long long n, int b) { return dim3((unsigned)((n + b - 1) / b)); }
static inline int padM(int m) { return (m + 63) & ~63; }
static inline int padK(int k) { return (k + 31) & ~31; }

struct GatBufs {
    float *XL, *XR, *EA, *LOG, *EXS, *MM, *DEN;
    unsigned short *ABF, *WBF;
};

static void run_gatv2(hipStream_t st,
                      const float* Xin, int Mn, int Kin,
                      const float* Eattr, int Ke, int nE,
                      const int* src, const int* dst,
                      const float* Wl, const float* bl,
                      const float* Wr, const float* br,
                      const float* We, const float* att, const float* bias,
                      int Hh, float* Out, bool relu, const GatBufs& B)
{
    const int HCs = Hh * HIDC;
    const int Mp = padM(Mn), Kp = padK(Kin);
    const int Ep = padM(nE), Kep = padK(Ke);
    const long long nX  = (long long)Mp * Kp;
    const long long nEa = (long long)Ep * Kep;
    dim3 ggX(Mp / 64, HCs / 128);
    dim3 ggE(Ep / 64, HCs / 128);

    // XL = Xin @ Wl + bl ; XR = Xin @ Wr + br
    convert_pad_kernel<<<g1(nX, 256), 256, 0, st>>>(Xin, B.ABF, Mn, Kin, Kp, nX);
    wt_convert_pad_kernel<<<g1((long long)Kp * HCs, 256), 256, 0, st>>>(Wl, B.WBF, Kin, Kp, HCs);
    gemm_bf16_wmma<<<ggX, 256, 0, st>>>(B.ABF, B.WBF, bl, B.XL, Kp, HCs);
    wt_convert_pad_kernel<<<g1((long long)Kp * HCs, 256), 256, 0, st>>>(Wr, B.WBF, Kin, Kp, HCs);
    gemm_bf16_wmma<<<ggX, 256, 0, st>>>(B.ABF, B.WBF, br, B.XR, Kp, HCs);

    // EA = Eattr @ We
    convert_pad_kernel<<<g1(nEa, 256), 256, 0, st>>>(Eattr, B.ABF, nE, Ke, Kep, nEa);
    wt_convert_pad_kernel<<<g1((long long)Kep * HCs, 256), 256, 0, st>>>(We, B.WBF, Ke, Kep, HCs);
    gemm_bf16_wmma<<<ggE, 256, 0, st>>>(B.ABF, B.WBF, nullptr, B.EA, Kep, HCs);

    // attention: segment-max -> exp -> segment-sum -> scatter
    const long long tasks = (long long)nE * Hh;
    fill_kernel<<<g1((long long)Mn * Hh, 256), 256, 0, st>>>(B.MM, -INFINITY, (long long)Mn * Hh);
    edge_logits_kernel<<<g1(tasks, 8), 256, 0, st>>>(B.XL, B.XR, B.EA, src, dst, att,
                                                     B.LOG, B.MM, nE, Hh);
    fill_kernel<<<g1((long long)Mn * Hh, 256), 256, 0, st>>>(B.DEN, 0.0f, (long long)Mn * Hh);
    edge_exp_kernel<<<g1(tasks, 256), 256, 0, st>>>(B.LOG, dst, B.MM, B.EXS, B.DEN, nE, Hh);
    fill_kernel<<<g1((long long)Mn * HCs, 256), 256, 0, st>>>(Out, 0.0f, (long long)Mn * HCs);
    edge_scatter_kernel<<<g1(tasks, 8), 256, 0, st>>>(B.XL, src, dst, B.EXS, B.DEN, Out, nE, Hh);
    finalize_kernel<<<g1((long long)Mn * HCs, 256), 256, 0, st>>>(Out, bias, HCs,
                                                                  (long long)Mn * HCs, relu ? 1 : 0);
}

extern "C" void kernel_launch(void* const* d_in, const int* in_sizes, int n_in,
                              void* d_out, int out_size, void* d_ws, size_t ws_size,
                              hipStream_t stream)
{
    (void)in_sizes; (void)n_in; (void)out_size; (void)ws_size;

    const float* x          = (const float*)d_in[0];   // [N,128]
    const int*   bond_index = (const int*)  d_in[1];   // [2,E]
    const float* bond_attr  = (const float*)d_in[2];   // [E,64]
    const int*   angle_index= (const int*)  d_in[3];   // [2,A]
    const float* angle_attr = (const float*)d_in[4];   // [A,16]
    const int*   batch      = (const int*)  d_in[5];   // [N]

    // params flattened in dict order: 5 convs x {Wl,bl,Wr,br,We,att,bias}, then gamma,beta
    const float* P[35];
    for (int i = 0; i < 35; ++i) P[i] = (const float*)d_in[6 + i];
    const float* bn_gamma = (const float*)d_in[41];
    const float* bn_beta  = (const float*)d_in[42];
    #define CP(c, k) P[(c) * 7 + (k)]   // k: 0 Wl,1 bl,2 Wr,3 br,4 We,5 att,6 bias

    const int* bsrc = bond_index;             const int* bdst = bond_index + N_EDGES;
    const int* asrc = angle_index;            const int* adst = angle_index + N_ANGLES;

    // ---- workspace arena (padded rows: E->200000, A->300032, N->50048) ----
    char* p = (char*)d_ws;
    auto alloc = [&](size_t bytes) -> void* {
        void* r = (void*)p; p += (bytes + 255) & ~(size_t)255; return r;
    };
    const size_t EP = (size_t)padM(N_EDGES), AP = (size_t)padM(N_ANGLES);
    GatBufs B;
    B.XL  = (float*)alloc(EP * 512 * 4);                 // max Mpad = Epad, max HCs = 512
    B.XR  = (float*)alloc(EP * 512 * 4);
    B.EA  = (float*)alloc(AP * 512 * 4);                 // max nEpad = Apad
    B.ABF = (unsigned short*)alloc(EP * 512 * 2);
    B.WBF = (unsigned short*)alloc((size_t)512 * 512 * 2);
    B.LOG = (float*)alloc((size_t)N_ANGLES * 4 * 4);
    B.EXS = (float*)alloc((size_t)N_ANGLES * 4 * 4);
    B.MM  = (float*)alloc((size_t)N_EDGES  * 4 * 4);
    B.DEN = (float*)alloc((size_t)N_EDGES  * 4 * 4);
    float* H      = (float*)alloc((size_t)N_NODES * 512 * 4);
    float* BA     = (float*)alloc((size_t)N_EDGES * 512 * 4);
    float* H3     = (float*)alloc((size_t)N_NODES * 128 * 4);
    float* POOLED = (float*)alloc((size_t)N_GRAPHS * 128 * 4);
    float* CNT    = (float*)alloc((size_t)N_GRAPHS * 4);

    // ---- pipeline ----
    // h = relu(conv1(x, bond, bond_attr))
    run_gatv2(stream, x, N_NODES, 128, bond_attr, 64, N_EDGES, bsrc, bdst,
              CP(0,0), CP(0,1), CP(0,2), CP(0,3), CP(0,4), CP(0,5), CP(0,6),
              4, H, true, B);
    // ba = Hconv1(bond_attr, angle, angle_attr)
    run_gatv2(stream, bond_attr, N_EDGES, 64, angle_attr, 16, N_ANGLES, asrc, adst,
              CP(1,0), CP(1,1), CP(1,2), CP(1,3), CP(1,4), CP(1,5), CP(1,6),
              4, BA, false, B);
    // h = conv2(h, bond, ba)  (relu folded: reference applies relu(h) before conv3)
    run_gatv2(stream, H, N_NODES, 512, BA, 512, N_EDGES, bsrc, bdst,
              CP(2,0), CP(2,1), CP(2,2), CP(2,3), CP(2,4), CP(2,5), CP(2,6),
              4, H, true, B);
    // ba = Hconv2(ba, angle, angle_attr)  (in-place safe: ba consumed via bf16/XL first)
    run_gatv2(stream, BA, N_EDGES, 512, angle_attr, 16, N_ANGLES, asrc, adst,
              CP(3,0), CP(3,1), CP(3,2), CP(3,3), CP(3,4), CP(3,5), CP(3,6),
              4, BA, false, B);
    // h = relu(conv3(h, bond, ba)) , heads=1
    run_gatv2(stream, H, N_NODES, 512, BA, 512, N_EDGES, bsrc, bdst,
              CP(4,0), CP(4,1), CP(4,2), CP(4,3), CP(4,4), CP(4,5), CP(4,6),
              1, H3, true, B);

    // global mean pool + BatchNorm (batch statistics)
    fill_kernel<<<g1(N_GRAPHS, 256), 256, 0, stream>>>(CNT, 0.0f, N_GRAPHS);
    fill_kernel<<<g1((long long)N_GRAPHS * 128, 256), 256, 0, stream>>>(
        POOLED, 0.0f, (long long)N_GRAPHS * 128);
    count_kernel<<<g1(N_NODES, 256), 256, 0, stream>>>(batch, CNT, N_NODES);
    pool_kernel<<<g1((long long)N_NODES * 128, 256), 256, 0, stream>>>(H3, batch, POOLED, N_NODES);
    bn_kernel<<<dim3(128), 256, 0, stream>>>(POOLED, CNT, bn_gamma, bn_beta,
                                             (float*)d_out, N_GRAPHS);
}
